// DepthConv_1932735283387
// MI455X (gfx1250) — matrix operand order repaired and checked by reading, not compiled, not run on hardware
//
#include <hip/hip_runtime.h>

typedef __attribute__((ext_vector_type(16))) _Float16 v16h;
typedef __attribute__((ext_vector_type(8)))  float    v8f;

#define ALPHA 8.3f
#define CIN   512
#define COUT  64
#define HH    64
#define WW    64
#define BB    8
#define NTAPS 9
#define KTOT   (CIN * NTAPS)      // 4608
#define KSTEPS (KTOT / 32)        // 144
#define WH_HALFS (KSTEPS * 4 * 32 * 16)  // 294912 halfs = 576 KB
#define XT_BYTE_OFF (1u << 20)           // xt region starts 1 MB into ws

// ---------------------------------------------------------------------------
// Kernel 1: weight (64,512,3,3) f32 -> f16, pre-swizzled into WMMA A-fragment
// order. Buffer layout: [ks][mt][lane][e] so the main kernel loads one
// contiguous v16h (32B) per lane. K ordering is tap-major: k = t*512 + c.
// 16-bit A 16x32 layout (contiguous k order): element e of lane holds
//   K = k0 + (lane/16)*16 + e,  M = mt*16 + lane%16.
// ---------------------------------------------------------------------------
__global__ __launch_bounds__(256) void wfrag_kernel(
    const float* __restrict__ w, _Float16* __restrict__ wh) {
  int tid = blockIdx.x * 256 + threadIdx.x;
  if (tid >= WH_HALFS) return;
  int e    = tid & 15;
  int lane = (tid >> 4) & 31;
  int mt   = (tid >> 9) & 3;
  int ks   = tid >> 11;
  int hi   = lane >> 4;
  int k = ks * 32 + hi * 16 + e;
  int m = mt * 16 + (lane & 15);
  int t = k / CIN;           // tap index = ki*3 + kj
  int c = k % CIN;
  wh[tid] = (_Float16)w[((size_t)m * CIN + c) * 9 + t];
}

// ---------------------------------------------------------------------------
// Kernel 2: x (B,C,H,W) f32 -> xt (B,H,W,C) f16.  LDS-tiled transpose:
// block = (b, h, 32-channel chunk); coalesced f32 reads, 64B-contiguous f16
// writes; LDS pitch 66 halfs to avoid bank conflicts on the transposed read.
// ---------------------------------------------------------------------------
__global__ __launch_bounds__(256) void xt_kernel(
    const float* __restrict__ x, _Float16* __restrict__ xt) {
  __shared__ _Float16 lds[32 * 66];
  int blk = blockIdx.x;                 // 8 * 64 * 16 = 8192 blocks
  int cb  = blk & 15;
  int h   = (blk >> 4) & 63;
  int b   = blk >> 10;
  int tid = threadIdx.x;
#pragma unroll
  for (int i = 0; i < 8; ++i) {
    int e   = tid + i * 256;
    int c32 = e >> 6;                   // 0..31
    int wcol = e & 63;                  // 0..63
    lds[c32 * 66 + wcol] =
        (_Float16)x[(((size_t)b * CIN + cb * 32 + c32) * HH + h) * WW + wcol];
  }
  __syncthreads();
#pragma unroll
  for (int i = 0; i < 8; ++i) {
    int e    = tid + i * 256;
    int wcol = e >> 5;                  // 0..63
    int c32  = e & 31;                  // 0..31
    xt[(((size_t)b * HH + h) * WW + wcol) * CIN + cb * 32 + c32] =
        lds[c32 * 66 + wcol];
  }
}

// ---------------------------------------------------------------------------
// Main kernel: implicit GEMM, M=64 (Cout), N=16 pixels/wave, K=4608.
// Block = 256 threads = 8 waves covering (b, h0..h0+1) x all 64 w.
// Wave: 4 accumulator subtiles (full Cout) x 16 consecutive w.
// Per tap: 16 K-steps of V_WMMA_F32_16X16X32_F16, then out += gate * tap_acc.
// ---------------------------------------------------------------------------
__global__ __launch_bounds__(256) void depthconv_kernel(
    const _Float16* __restrict__ xt, const _Float16* __restrict__ wh,
    const float* __restrict__ depth, float* __restrict__ out) {
  int blk  = blockIdx.x;                // 256 blocks
  int b    = blk >> 5;
  int h0   = (blk & 31) * 2;
  int wi   = threadIdx.x >> 5;          // wave id 0..7
  int lane = threadIdx.x & 31;
  int h    = h0 + (wi >> 2);
  int w0   = (wi & 3) * 16;
  int n    = lane & 15;                 // output column (pixel) in tile
  int hi   = lane >> 4;
  int wpix = w0 + n;

  const float* dimg = depth + (size_t)b * (HH * WW);
  float d0 = dimg[h * WW + wpix];

  v8f acc[4] = {};

  for (int t = 0; t < NTAPS; ++t) {
    int di = t / 3 - 1, dj = t % 3 - 1;
    int hs = h + di;
    int ws = wpix + dj;
    bool valid = ((unsigned)hs < HH) && ((unsigned)ws < WW);
    int hc = min(max(hs, 0), HH - 1);
    int wc = min(max(ws, 0), WW - 1);
    float gate = valid ? __expf(-ALPHA * fabsf(d0 - dimg[hc * WW + wc])) : 0.0f;

    // B fragment base: 16 consecutive channels for this lane's source pixel
    const _Float16* bp = xt + (((size_t)b * HH + hc) * WW + wc) * CIN + hi * 16;
    // A fragment base for ks = t*16
    const _Float16* ap = wh + ((size_t)(t * 16) * 4) * 32 * 16 + lane * 16;

    v8f tap[4] = {};
#pragma unroll 2
    for (int ccs = 0; ccs < 16; ++ccs) {
      v16h bfrag = *(const v16h*)(bp + ccs * 32);
      const _Float16* apc = ap + (size_t)ccs * 4 * 32 * 16;
#pragma unroll
      for (int mt = 0; mt < 4; ++mt) {
        v16h afrag = *(const v16h*)(apc + mt * 32 * 16);
        tap[mt] = __builtin_amdgcn_wmma_f32_16x16x32_f16(
            false, afrag, false, bfrag, (short)0, tap[mt], false, false);
      }
    }
#pragma unroll
    for (int mt = 0; mt < 4; ++mt) acc[mt] += gate * tap[mt];
  }

  // Store: lane holds column n; VGPR r holds rows m = mt*16 + r + hi*8.
#pragma unroll
  for (int mt = 0; mt < 4; ++mt) {
#pragma unroll
    for (int r = 0; r < 8; ++r) {
      int m = mt * 16 + r + hi * 8;
      out[(((size_t)b * COUT + m) * HH + h) * WW + wpix] = acc[mt][r];
    }
  }
}

// ---------------------------------------------------------------------------
extern "C" void kernel_launch(void* const* d_in, const int* in_sizes, int n_in,
                              void* d_out, int out_size, void* d_ws, size_t ws_size,
                              hipStream_t stream) {
  const float* x      = (const float*)d_in[0];  // (8,512,64,64)
  const float* depth  = (const float*)d_in[1];  // (8,1,64,64)
  const float* weight = (const float*)d_in[2];  // (64,512,3,3)
  float* out = (float*)d_out;                   // (8,64,64,64)

  _Float16* wh = (_Float16*)d_ws;                           // 576 KB
  _Float16* xt = (_Float16*)((char*)d_ws + XT_BYTE_OFF);    // 32 MB

  wfrag_kernel<<<(WH_HALFS + 255) / 256, 256, 0, stream>>>(weight, wh);
  xt_kernel<<<BB * HH * (CIN / 32), 256, 0, stream>>>(x, xt);
  depthconv_kernel<<<BB * (HH / 2), 256, 0, stream>>>(xt, wh, depth, out);
}
